// GCN_74113955659885
// MI455X (gfx1250) — compile-verified, matrix operand order checked by reading, not run on hardware
//
#include <hip/hip_runtime.h>

#define HID 64

typedef __attribute__((ext_vector_type(16))) __bf16 v16bf;
typedef __attribute__((ext_vector_type(8)))  __bf16 v8bf;
typedef __attribute__((ext_vector_type(8)))  float  v8f;

// ---------------- degree / norm ----------------
__global__ void k_deg_init(float* __restrict__ deg, int n) {
    int i = blockIdx.x * blockDim.x + threadIdx.x;
    if (i < n) deg[i] = 1.0f;   // self-loop contributes 1 to every node's degree
}

__global__ void k_deg_count(const int* __restrict__ dst, float* __restrict__ deg, int E) {
    int e = blockIdx.x * blockDim.x + threadIdx.x;
    if (e < E) atomicAdd(&deg[dst[e]], 1.0f);
}

__global__ void k_deg_rsqrt(float* __restrict__ dis, int n) {
    int i = blockIdx.x * blockDim.x + threadIdx.x;
    if (i < n) dis[i] = rsqrtf(dis[i]);   // deg > 0 always (self loops)
}

// ---------------- bf16 WMMA GEMM: OUT[M,64] = X[M,K] * W[K,64] ----------------
// One wave -> 16 rows x 64 cols. Block = 256 threads = 8 waves = 128 rows.
__global__ __launch_bounds__(256)
void k_gemm_bf16(const float* __restrict__ X, const float* __restrict__ Wg,
                 float* __restrict__ OUT, int M, int K) {
    __shared__ __bf16 sW[256 * HID];           // transposed: sW[n*K + k], 32 KB max
    const int tid = threadIdx.x;
    for (int idx = tid; idx < K * HID; idx += 256) {
        int k = idx >> 6, n = idx & 63;        // Wg row-major [K,64]
        sW[n * K + k] = (__bf16)Wg[idx];
    }
    __syncthreads();

    const int lane = tid & 31;
    const int wave = tid >> 5;                 // wave32
    const int r0   = blockIdx.x * 128 + wave * 16;
    if (r0 >= M) return;                       // wave-uniform: EXEC stays all-1s

    const int m = lane & 15;                   // row within tile / col within B tile
    const int g = lane >> 4;                   // K-half selector

    v8f acc[4] = {};                           // 4 tiles of 16x16 f32 (N = 64)

    const float*  xrow = X + (size_t)(r0 + m) * K + g * 8;
    const __bf16* wcol = sW + (size_t)m * K + g * 8;   // tile nt adds 16*K

    #pragma unroll 2
    for (int k0 = 0; k0 < K; k0 += 32) {
        // ---- load phase: A (global, f32->bf16) + all 4 B tiles (LDS) ----
        float4 a0 = *(const float4*)(xrow + k0);
        float4 a1 = *(const float4*)(xrow + k0 + 4);
        float4 a2 = *(const float4*)(xrow + k0 + 16);
        float4 a3 = *(const float4*)(xrow + k0 + 20);

        v8bf blo[4], bhi[4];
        #pragma unroll
        for (int nt = 0; nt < 4; ++nt) {
            const __bf16* p = wcol + (size_t)nt * 16 * K + k0;
            blo[nt] = *(const v8bf*)(p);
            bhi[nt] = *(const v8bf*)(p + 16);
        }

        v16bf a;
        a[0] = (__bf16)a0.x; a[1] = (__bf16)a0.y; a[2]  = (__bf16)a0.z; a[3]  = (__bf16)a0.w;
        a[4] = (__bf16)a1.x; a[5] = (__bf16)a1.y; a[6]  = (__bf16)a1.z; a[7]  = (__bf16)a1.w;
        a[8] = (__bf16)a2.x; a[9] = (__bf16)a2.y; a[10] = (__bf16)a2.z; a[11] = (__bf16)a2.w;
        a[12]= (__bf16)a3.x; a[13]= (__bf16)a3.y; a[14] = (__bf16)a3.z; a[15] = (__bf16)a3.w;

        // ---- compute phase: 4 back-to-back WMMAs (no per-tile LDS wait) ----
        #pragma unroll
        for (int nt = 0; nt < 4; ++nt) {
            v16bf b;
            #pragma unroll
            for (int j = 0; j < 8; ++j) { b[j] = blo[nt][j]; b[j + 8] = bhi[nt][j]; }
            acc[nt] = __builtin_amdgcn_wmma_f32_16x16x32_bf16(
                false, a, false, b, (short)0, acc[nt], false, false);
        }
    }

    // C/D layout: VGPR j -> row r0 + j + 8*g, col = nt*16 + m
    #pragma unroll
    for (int nt = 0; nt < 4; ++nt) {
        float* op = OUT + (size_t)(r0 + 8 * g) * HID + nt * 16 + m;
        #pragma unroll
        for (int j = 0; j < 8; ++j) op[(size_t)j * HID] = acc[nt][j];
    }
}

// ---------------- aggregation ----------------
// self-loop term (also initializes the accumulator buffer): out = h * dis^2
__global__ void k_agg_self(const float* __restrict__ h, const float* __restrict__ dis,
                           float* __restrict__ out, int n) {
    long long idx = (long long)blockIdx.x * blockDim.x + threadIdx.x;
    if (idx >= (long long)n * HID) return;
    int i = (int)(idx >> 6);
    float d = dis[i];
    out[idx] = h[idx] * d * d;
}

// one wave per edge, 2 feats per lane; scatter-add lives in L2 (out = 25.6 MB)
__global__ void k_agg_edges(const float* __restrict__ h, const float* __restrict__ dis,
                            const int* __restrict__ src, const int* __restrict__ dst,
                            float* __restrict__ out, int E) {
    long long t = (long long)blockIdx.x * blockDim.x + threadIdx.x;
    long long e = t >> 5;
    if (e >= E) return;
    int f2 = (int)(t & 31) * 2;
    int s = src[e], d = dst[e];
    float w = dis[s] * dis[d];
    float2 hv = *(const float2*)(h + (size_t)s * HID + f2);
    atomicAdd(out + (size_t)d * HID + f2,     hv.x * w);
    atomicAdd(out + (size_t)d * HID + f2 + 1, hv.y * w);
}

__global__ void k_bias_relu(float* __restrict__ out, const float* __restrict__ b, int n) {
    long long idx = (long long)blockIdx.x * blockDim.x + threadIdx.x;
    if (idx >= (long long)n * HID) return;
    int f = (int)(idx & 63);
    out[idx] = fmaxf(out[idx] + b[f], 0.0f);
}

// ---------------- classifier head: out[N,6] = h[N,64] @ Wfc[64,6] + bfc ----------------
__global__ void k_fc(const float* __restrict__ h, const float* __restrict__ Wfc,
                     const float* __restrict__ bfc, float* __restrict__ out, int n) {
    int i = blockIdx.x * blockDim.x + threadIdx.x;
    if (i >= n) return;
    const float* hr = h + (size_t)i * HID;
    float acc[6];
    #pragma unroll
    for (int c = 0; c < 6; ++c) acc[c] = bfc[c];
    for (int f = 0; f < HID; ++f) {
        float hv = hr[f];
        #pragma unroll
        for (int c = 0; c < 6; ++c) acc[c] += hv * Wfc[f * 6 + c];
    }
    #pragma unroll
    for (int c = 0; c < 6; ++c) out[(size_t)i * 6 + c] = acc[c];
}

// ---------------- driver ----------------
extern "C" void kernel_launch(void* const* d_in, const int* in_sizes, int n_in,
                              void* d_out, int out_size, void* d_ws, size_t ws_size,
                              hipStream_t stream) {
    (void)n_in; (void)out_size; (void)ws_size;
    const float* x   = (const float*)d_in[0];
    const int*   ei  = (const int*)d_in[1];
    const float* W1  = (const float*)d_in[2];
    const float* b1  = (const float*)d_in[3];
    const float* W2  = (const float*)d_in[4];
    const float* b2  = (const float*)d_in[5];
    const float* Wfc = (const float*)d_in[6];
    const float* bfc = (const float*)d_in[7];
    float* out = (float*)d_out;

    const int N = in_sizes[0] / 256;   // 100000 nodes
    const int E = in_sizes[1] / 2;     // 3200000 edges
    const int* src = ei;               // edge_index[0]
    const int* dst = ei + E;           // edge_index[1]

    char* ws = (char*)d_ws;
    float* dis  = (float*)ws;
    size_t off  = ((size_t)N * 4 + 255) & ~(size_t)255;
    float* bufA = (float*)(ws + off);
    float* bufB = bufA + (size_t)N * HID;

    const unsigned gN  = (unsigned)((N + 255) / 256);
    const unsigned gE  = (unsigned)((E + 255) / 256);
    const unsigned gNF = (unsigned)(((size_t)N * HID + 255) / 256);
    const unsigned gEW = (unsigned)(((size_t)E * 32 + 255) / 256);
    const unsigned gMM = (unsigned)((N + 127) / 128);

    // symmetric normalization: dis = rsqrt(1 + in-degree)
    k_deg_init <<<gN,  256, 0, stream>>>(dis, N);
    k_deg_count<<<gE,  256, 0, stream>>>(dst, dis, E);
    k_deg_rsqrt<<<gN,  256, 0, stream>>>(dis, N);

    // layer 1: h1 = relu(norm-agg(x @ W1) + b1)
    k_gemm_bf16<<<gMM, 256, 0, stream>>>(x, W1, bufA, N, 256);
    k_agg_self <<<gNF, 256, 0, stream>>>(bufA, dis, bufB, N);
    k_agg_edges<<<gEW, 256, 0, stream>>>(bufA, dis, src, dst, bufB, E);
    k_bias_relu<<<gNF, 256, 0, stream>>>(bufB, b1, N);

    // layer 2: h2 = relu(norm-agg(h1 @ W2) + b2)
    k_gemm_bf16<<<gMM, 256, 0, stream>>>(bufB, W2, bufA, N, HID);
    k_agg_self <<<gNF, 256, 0, stream>>>(bufA, dis, bufB, N);
    k_agg_edges<<<gEW, 256, 0, stream>>>(bufA, dis, src, dst, bufB, E);
    k_bias_relu<<<gNF, 256, 0, stream>>>(bufB, b2, N);

    // head
    k_fc<<<gN, 256, 0, stream>>>(bufB, Wfc, bfc, out, N);
}
